// PerturbedBlock_56341380989152
// MI455X (gfx1250) — matrix operand order repaired and checked by reading, not compiled, not run on hardware
//
#include <hip/hip_runtime.h>

typedef __attribute__((ext_vector_type(16))) _Float16 v16h;
typedef __attribute__((ext_vector_type(8)))  float    v8f;
typedef __attribute__((ext_vector_type(8)))  unsigned v8u;

#define BB 32
#define NN 64
#define HH 128
#define SS 100
#define CC 16
#define SBTOT (SS * BB)

__device__ __forceinline__ unsigned pack2(float lo, float hi) {
  union { _Float16 h[2]; unsigned u; } p;
  p.h[0] = (_Float16)lo;
  p.h[1] = (_Float16)hi;
  return p.u;
}

// ---------------------------------------------------------------------------
// Swizzled-B u32 index: B(KxN) f16 packed along K-pairs so one lane's 8 pairs
// for a 16x16x32 WMMA step are contiguous (2x b128 loads).
//   idx(k, n) = (k>>5)*(N*16) + n*16 + ((k>>4)&1)*8 + ((k&15)>>1)
// ---------------------------------------------------------------------------
__device__ __forceinline__ int bswz_idx(int k, int n, int ldbN) {
  return (k >> 5) * (ldbN * 16) + n * 16 + (((k >> 4) & 1) << 3) + ((k & 15) >> 1);
}

// ---------------------------------------------------------------------------
// One 16x16 f32 tile of C = A(MxK) @ B(KxN).
// A: f16 row-major (LDS/global), K-pairs contiguous -> 2x b128 per step.
// B: swizzled u32 layout -> 2x b128 per step.
// Lane mappings: row/col = lane%16; k = 16*(lane/16) + s in both A and B (contiguous order).
// ---------------------------------------------------------------------------
__device__ __forceinline__ v8f wmma_tile(const _Float16* As, int lda,
                                         const unsigned* Bswz, int ldbN,
                                         int K, int m0, int n0, int lane) {
  v8f acc = {0.f, 0.f, 0.f, 0.f, 0.f, 0.f, 0.f, 0.f};
  const int half_id = (lane >> 4) & 1;
  const int mrow = m0 + (lane & 15);
  const int ncol = n0 + (lane & 15);
  const unsigned* arow = (const unsigned*)As + ((mrow * lda) >> 1);
  for (int k0 = 0; k0 < K; k0 += 32) {
    const unsigned* ap = arow + (k0 >> 1);
    const unsigned* bp = Bswz + (k0 >> 5) * (ldbN * 16) + ncol * 16 + (half_id << 3);
    v8u aw, bw;
#pragma unroll
    for (int i = 0; i < 8; ++i) {
      aw[i] = ap[(half_id << 3) + i];
      bw[i] = bp[i];
    }
    acc = __builtin_amdgcn_wmma_f32_16x16x32_f16(
        false, __builtin_bit_cast(v16h, aw), false, __builtin_bit_cast(v16h, bw),
        (short)0, acc, false, false);
  }
  return acc;
}

// ---------------------------------------------------------------------------
// Convert weights f32 -> f16 swizzled-B u32 layout in workspace (once/launch).
// ---------------------------------------------------------------------------
__global__ void convert_weights_kernel(const float* W1, const float* W2,
                                       const float* cW1, const float* cW2,
                                       unsigned* W1s, unsigned* W2s,
                                       unsigned* cW1s, unsigned* cW2s) {
  const int t = blockIdx.x * 256 + threadIdx.x;  // K-pair major index
  if (t < (HH / 2) * HH) {
    const int k = (t / HH) * 2, n = t % HH;
    const int o = bswz_idx(k, n, HH);
    W1s[o]  = pack2(W1[k * HH + n],  W1[(k + 1) * HH + n]);
    W2s[o]  = pack2(W2[k * HH + n],  W2[(k + 1) * HH + n]);
    cW1s[o] = pack2(cW1[k * HH + n], cW1[(k + 1) * HH + n]);
  }
  if (t < (HH / 2) * CC) {
    const int k = (t / CC) * 2, n = t % CC;
    const int o = bswz_idx(k, n, CC);
    cW2s[o] = pack2(cW2[k * CC + n], cW2[(k + 1) * CC + n]);
  }
}

// Zero mean-accumulator output regions + per-batch cluster-count maxima.
__global__ void zero_kernel(float* out_accum, int* ncmax) {
  const int t = blockIdx.x * 256 + threadIdx.x;
  if (t < BB * NN * HH + BB * NN * NN) out_accum[t] = 0.f;
  if (t < BB) ncmax[t] = 0;
}

// ---------------------------------------------------------------------------
// Symmetric normalization of adjacency with forced self-loops. grid=B, blk=64.
// ---------------------------------------------------------------------------
__global__ void norm_adj_kernel(const float* adj, float* a_norm) {
  const int b = blockIdx.x, i = threadIdx.x;
  __shared__ float d[NN];
  const float* A = adj + b * NN * NN;
  float s = 0.f;
  for (int j = 0; j < NN; ++j) s += (j == i) ? 1.f : A[i * NN + j];
  s = (s < 1.f) ? 1.f : s;
  d[i] = 1.0f / sqrtf(s);
  __syncthreads();
  const float di = d[i];
  for (int j = 0; j < NN; ++j) {
    const float aij = (j == i) ? 1.f : A[i * NN + j];
    a_norm[b * NN * NN + i * NN + j] = di * aij * d[j];
  }
}

// ---------------------------------------------------------------------------
// One GCN layer fused per batch: out = relu(a_norm @ (x @ W) + bias) * mask
// grid = B, block = 256 (8 waves). W arrives pre-swizzled; h is written
// swizzled by GEMM1's epilogue (it is the B operand of GEMM2).
// ---------------------------------------------------------------------------
__global__ void gcn_layer_kernel(const float* xin, const unsigned* Ws,
                                 const float* bias, const float* a_norm,
                                 const int* mask, float* out) {
  const int b = blockIdx.x;
  const int tid = threadIdx.x;
  const int wave = tid >> 5, lane = tid & 31;
  __shared__ __align__(16) _Float16 xh[NN * HH];        // 16 KB, A of GEMM1
  __shared__ __align__(16) unsigned wsw[HH * HH / 2];   // 32 KB, B of GEMM1
  __shared__ __align__(16) unsigned hhs[NN * HH / 2];   // 16 KB, B of GEMM2
  __shared__ __align__(16) _Float16 ah[NN * NN];        // 8 KB,  A of GEMM2

  for (int t = tid; t < NN * HH; t += 256) xh[t] = (_Float16)xin[b * NN * HH + t];
  for (int t = tid; t < HH * HH / 8; t += 256)
    ((uint4*)wsw)[t] = ((const uint4*)Ws)[t];
  for (int t = tid; t < NN * NN; t += 256) ah[t] = (_Float16)a_norm[b * NN * NN + t];
  __syncthreads();

  const int half_id = (lane >> 4) & 1, ncol = lane & 15;

  // h = x @ W : 4x8 tiles, 4 per wave; epilogue writes swizzled (one b128)
  for (int q = 0; q < 4; ++q) {
    const int tI = wave * 4 + q;
    const int mt = tI >> 3, nt = tI & 7;
    v8f acc = wmma_tile(xh, HH, wsw, HH, HH, mt * 16, nt * 16, lane);
    // rows mt*16 + half_id*8 + r at col nt*16+ncol -> 4 consecutive u32
    const int base = (mt >> 1) * (HH * 16) + (nt * 16 + ncol) * 16 +
                     ((mt & 1) << 3) + (half_id << 2);
    uint4 w;
    w.x = pack2(acc[0], acc[1]);
    w.y = pack2(acc[2], acc[3]);
    w.z = pack2(acc[4], acc[5]);
    w.w = pack2(acc[6], acc[7]);
    *(uint4*)(hhs + base) = w;
  }
  __syncthreads();

  // out = a_norm @ h + bias, relu, mask
  for (int q = 0; q < 4; ++q) {
    const int tI = wave * 4 + q;
    const int mt = tI >> 3, nt = tI & 7;
    v8f acc = wmma_tile(ah, NN, hhs, HH, NN, mt * 16, nt * 16, lane);
#pragma unroll
    for (int r = 0; r < 8; ++r) {
      const int row = mt * 16 + half_id * 8 + r;
      const int col = nt * 16 + ncol;
      float v = acc[r] + bias[col];
      v = v > 0.f ? v : 0.f;
      if (!mask[b * NN + row]) v = 0.f;
      out[b * NN * HH + row * HH + col] = v;
    }
  }
}

// ---------------------------------------------------------------------------
// Fused per-sample kernel. grid = S*B, block = 256.
// ---------------------------------------------------------------------------
__global__ void sample_kernel(const float* x_emb, const float* noise,
                              const float* adj, const int* mask,
                              const unsigned* cW1s, const float* cb1,
                              const unsigned* cW2s, const float* cb2,
                              float* out_xnew, float* out_adjnew,
                              float* out_assign, int* ncmax) {
  const int samp = blockIdx.x;  // i = s*B + b (sample-major tiling)
  const int b = samp & (BB - 1);
  const int tid = threadIdx.x;
  const int wave = tid >> 5, lane = tid & 31;

  __shared__ __align__(16) float    xf[NN * HH];        // 32 KB, live throughout
  __shared__ __align__(16) _Float16 sA[NN * HH];        // x f16 -> logits(f32)
  __shared__ __align__(16) unsigned sWu[HH * HH / 2];   // cW1 swz -> pooled_x(f32)
  __shared__ __align__(16) _Float16 sH[NN * HH];        // hidden -> pooled_adj(f32)
  __shared__ int labels[NN];
  __shared__ int concepts[NN];
  __shared__ int present[NN + 1];
  __shared__ int rankp[NN];
  __shared__ int assign[NN];
  __shared__ int changed;
  __shared__ int ncl;

  // 1) x_all = x_emb[b] + 0.5*noise[samp]  (vectorized single read of noise)
  {
    const float4* xe4 = (const float4*)(x_emb + b * NN * HH);
    const float4* nz4 = (const float4*)(noise + (size_t)samp * NN * HH);
    for (int t = tid; t < NN * HH / 4; t += 256) {
      const float4 xv = xe4[t], nv = nz4[t];
      float4 v;
      v.x = xv.x + 0.5f * nv.x;
      v.y = xv.y + 0.5f * nv.y;
      v.z = xv.z + 0.5f * nv.z;
      v.w = xv.w + 0.5f * nv.w;
      ((float4*)xf)[t] = v;
      sA[4 * t + 0] = (_Float16)v.x;
      sA[4 * t + 1] = (_Float16)v.y;
      sA[4 * t + 2] = (_Float16)v.z;
      sA[4 * t + 3] = (_Float16)v.w;
    }
    for (int t = tid; t < HH * HH / 8; t += 256)
      ((uint4*)sWu)[t] = ((const uint4*)cW1s)[t];
  }
  __syncthreads();

  const int half_id = (lane >> 4) & 1, ncol = lane & 15;

  // 2) hidden = relu(x_all @ cW1 + cb1) -> f16 row-major (A of GEMM2)
  for (int q = 0; q < 4; ++q) {
    const int tI = wave * 4 + q;
    const int mt = tI >> 3, nt = tI & 7;
    v8f acc = wmma_tile(sA, HH, sWu, HH, HH, mt * 16, nt * 16, lane);
#pragma unroll
    for (int r = 0; r < 8; ++r) {
      const int row = mt * 16 + half_id * 8 + r;
      const int col = nt * 16 + ncol;
      const float v = acc[r] + cb1[col];
      sH[row * HH + col] = (_Float16)(v > 0.f ? v : 0.f);
    }
  }
  __syncthreads();

  // 3) logits = hidden @ cW2 + cb2 (B = swizzled cW2 straight from L2)
  float* logits = reinterpret_cast<float*>(sA);
  if (wave < 4) {
    v8f acc = wmma_tile(sH, HH, cW2s, CC, HH, wave * 16, 0, lane);
#pragma unroll
    for (int r = 0; r < 8; ++r) {
      const int row = wave * 16 + half_id * 8 + r;
      logits[row * CC + ncol] = acc[r] + cb2[ncol];
    }
  }
  __syncthreads();

  // 4) argmax over C=16 (first-max tie break, matching jnp.argmax)
  if (tid < NN) {
    float best = logits[tid * CC];
    int bi = 0;
    for (int j = 1; j < CC; ++j) {
      const float v = logits[tid * CC + j];
      if (v > best) { best = v; bi = j; }
    }
    concepts[tid] = bi;
  }
  __syncthreads();

  // 5) neighbor bitmask: edge = adj>0 & same-concept & mask_i & mask_j
  unsigned long long nb = 0ull;
  if (tid < NN) {
    const bool mi = mask[b * NN + tid] != 0;
    const int ci = concepts[tid];
    if (mi) {
      for (int j = 0; j < NN; ++j) {
        if (adj[b * NN * NN + tid * NN + j] > 0.f && mask[b * NN + j] != 0 &&
            concepts[j] == ci)
          nb |= (1ull << j);
      }
    }
    labels[tid] = mi ? tid : NN;
  }
  __syncthreads();

  // 6) min-label propagation (same fixpoint as 64 fixed iterations)
  for (int it = 0; it < NN; ++it) {
    int newl = 0;
    if (tid < NN) {
      newl = labels[tid];
      unsigned long long m = nb;
      while (m) {
        const int j = __builtin_ctzll(m);
        m &= (m - 1);
        const int lj = labels[j];
        newl = lj < newl ? lj : newl;
      }
    }
    __syncthreads();
    if (tid == 0) changed = 0;
    __syncthreads();
    if (tid < NN && newl != labels[tid]) { labels[tid] = newl; changed = 1; }
    __syncthreads();
    if (!changed) break;  // uniform across block
  }

  // 7) compact relabel (rank of root among present roots)
  if (tid < NN + 1) present[tid] = 0;
  __syncthreads();
  if (tid < NN && mask[b * NN + tid] != 0) present[labels[tid]] = 1;
  __syncthreads();
  if (tid == 0) {
    int c = 0;
    for (int n = 0; n < NN; ++n) { c += present[n]; rankp[n] = c; }
    ncl = c;  // num_clusters == max assignment
  }
  __syncthreads();
  if (tid < NN) {
    const int root = labels[tid] < (NN - 1) ? labels[tid] : (NN - 1);
    const int a = (mask[b * NN + tid] != 0) ? rankp[root] : 0;
    assign[tid] = a;
    out_assign[(size_t)samp * NN + tid] = (float)a;
  }
  if (tid == 0) atomicMax(&ncmax[b], ncl);
  __syncthreads();

  // 8) pooled x / adj in LDS (reuse sWu / sH as f32)
  float* px = reinterpret_cast<float*>(sWu);  // 64x128 f32 = 32 KB
  float* pa = reinterpret_cast<float*>(sH);   // 64x64  f32 = 16 KB
  for (int t = tid; t < NN * HH; t += 256) px[t] = 0.f;
  for (int t = tid; t < NN * NN; t += 256) pa[t] = 0.f;
  __syncthreads();
  for (int t = tid; t < NN * HH; t += 256) {
    const int n = t >> 7, h = t & (HH - 1);
    const int a = assign[n];
    if (a) atomicAdd(&px[(a - 1) * HH + h], xf[t]);
  }
  for (int t = tid; t < NN * NN; t += 256) {
    const int n = t >> 6, j = t & (NN - 1);
    const int an = assign[n], aj = assign[j];
    if (an && aj && adj[b * NN * NN + t] > 0.f)
      pa[(an - 1) * NN + (aj - 1)] = 1.0f;  // segment_max of binary adj
  }
  __syncthreads();

  // 9) mean-accumulate into outputs
  const float inv = 1.0f / (float)SS;
  for (int t = tid; t < NN * HH; t += 256) {
    const float v = px[t];
    if (v != 0.f) atomicAdd(&out_xnew[b * NN * HH + t], v * inv);
  }
  for (int t = tid; t < NN * NN; t += 256) {
    const float v = pa[t];
    if (v != 0.f) atomicAdd(&out_adjnew[b * NN * NN + t], v * inv);
  }
}

__global__ void finalize_mask_kernel(const int* ncmax, float* out_mask) {
  const int t = blockIdx.x * 256 + threadIdx.x;
  if (t < BB * NN) {
    const int b = t >> 6, n = t & (NN - 1);
    out_mask[t] = (n < ncmax[b]) ? 1.f : 0.f;
  }
}

// ---------------------------------------------------------------------------
extern "C" void kernel_launch(void* const* d_in, const int* in_sizes, int n_in,
                              void* d_out, int out_size, void* d_ws, size_t ws_size,
                              hipStream_t stream) {
  const float* x     = (const float*)d_in[0];
  const float* adj   = (const float*)d_in[1];
  const int*   mask  = (const int*)d_in[2];
  const float* W1    = (const float*)d_in[3];
  const float* b1    = (const float*)d_in[4];
  const float* W2    = (const float*)d_in[5];
  const float* b2    = (const float*)d_in[6];
  const float* cW1   = (const float*)d_in[7];
  const float* cb1   = (const float*)d_in[8];
  const float* cW2   = (const float*)d_in[9];
  const float* cb2   = (const float*)d_in[10];
  const float* noise = (const float*)d_in[11];
  // d_in[12] = num_samples (== SS, compile-time constant here)

  float* out        = (float*)d_out;
  float* out_xnew   = out;                                // 262144
  float* out_adjnew = out + BB * NN * HH;                 // +131072
  float* out_assign = out + BB * NN * HH + BB * NN * NN;  // +204800
  float* out_xemb   = out_assign + (size_t)SBTOT * NN;    // +262144
  float* out_mask   = out_xemb + BB * NN * HH;            // +2048

  char* ws = (char*)d_ws;
  unsigned* W1s   = (unsigned*)(ws);
  unsigned* W2s   = (unsigned*)(ws + 32768);
  unsigned* cW1s  = (unsigned*)(ws + 65536);
  unsigned* cW2s  = (unsigned*)(ws + 98304);
  float*    a_nrm = (float*)(ws + 102400);   // 32*64*64*4 = 524288
  float*    xe    = (float*)(ws + 626688);   // 32*64*128*4 = 1048576
  int*      ncmax = (int*)(ws + 1675264);    // 32 ints

  convert_weights_kernel<<<(HH * HH / 2 + 255) / 256, 256, 0, stream>>>(
      W1, W2, cW1, cW2, W1s, W2s, cW1s, cW2s);
  zero_kernel<<<(BB * NN * HH + BB * NN * NN + 255) / 256, 256, 0, stream>>>(
      out_xnew, ncmax);
  norm_adj_kernel<<<BB, NN, 0, stream>>>(adj, a_nrm);
  gcn_layer_kernel<<<BB, 256, 0, stream>>>(x, W1s, b1, a_nrm, mask, xe);
  gcn_layer_kernel<<<BB, 256, 0, stream>>>(xe, W2s, b2, a_nrm, mask, out_xemb);
  sample_kernel<<<SBTOT, 256, 0, stream>>>(out_xemb, noise, adj, mask, cW1s, cb1,
                                           cW2s, cb2, out_xnew, out_adjnew,
                                           out_assign, ncmax);
  finalize_mask_kernel<<<(BB * NN + 255) / 256, 256, 0, stream>>>(ncmax, out_mask);
}
